// LSTMModel_60129542324
// MI455X (gfx1250) — matrix operand order repaired and checked by reading, not compiled, not run on hardware
//
#include <hip/hip_runtime.h>

// ---------------------------------------------------------------------------
// LSTM (S=4096, I=64, H=1024, O=1) as a single persistent wave32 kernel.
//
// MI455X strategy:
//  * 32 persistent workgroups x 256 threads (8 waves). WG g owns hidden units
//    j in [g*32, g*32+32) -> 128 rows of W_hh / W_ih (4 gates x 32).
//  * Each wave keeps its 16x1024 W_hh slice resident in VGPRs as 32 bf16
//    WMMA B-fragments (256 VGPRs/lane) -> zero weight traffic per step.
//  * A-fragment (h in row M=0 only): every lane does two UNCONDITIONAL
//    ds_load_b128; lanes that must be zero read a 64B zeroed LDS block with
//    per-lane stride 0. No EXEC masking, no zero-fill per chunk.
//  * Inner loop: 2-deep software pipeline on A fragments (hide LDS latency)
//    and TWO accumulator chains (break the WMMA D->C serial dependence).
//  * h broadcast (2KB bf16) staged via GLOBAL_LOAD_ASYNC_TO_LDS_B64.
//  * Cross-WG sync per step: s_cluster_barrier (HW fast path if clustered,
//    NOP otherwise) + sense-reversing agent-scope atomic barrier.
// ---------------------------------------------------------------------------

typedef __attribute__((ext_vector_type(16))) __bf16 v16bf;
typedef __attribute__((ext_vector_type(8)))  float  v8f;
typedef __attribute__((ext_vector_type(2)))  int    v2i;

#define SEQ 4096
#define IDIM 64
#define HDIM 1024
#define NWG  32
#define TPB  256
#define JPW  (HDIM / NWG)   // 32 hidden units per WG

#if __has_builtin(__builtin_amdgcn_global_load_async_to_lds_b64) && \
    __has_builtin(__builtin_amdgcn_s_wait_asynccnt)
#define USE_ASYNC_LDS 1
typedef __attribute__((address_space(1))) v2i* gv2i_p;   // global int2*
typedef __attribute__((address_space(3))) v2i* lv2i_p;   // LDS int2*
#endif

struct Bar { unsigned cnt; unsigned gen; };

// Native bf16 convert (RNE on gfx1250 bf16 hardware).
__device__ __forceinline__ unsigned short f2bf(float f) {
  return __builtin_bit_cast(unsigned short, (__bf16)f);
}

// A-fragment load: two b128 LDS reads fully populate the 8-dword fragment.
// Element layout (bf16 16x32 A): e0-15 <-> K base..+15.
__device__ __forceinline__ v16bf load_afrag(const unsigned short* p) {
  v16bf a;
  const uint4* q = reinterpret_cast<const uint4*>(p);
  uint4* d = reinterpret_cast<uint4*>(&a);
  d[0] = q[0];        // bytes +0  .. +15
  d[1] = q[1];        // bytes +16 .. +31
  return a;
}

__device__ __forceinline__ void global_barrier(Bar* b, int tid) {
  __syncthreads();
  if (tid == 0) {
    // HW cluster barrier: fast path when dispatched as a cluster; ISA
    // guarantees NOP when ClusterID==0, so always safe.
    __builtin_amdgcn_s_cluster_barrier();
    unsigned g   = __hip_atomic_load(&b->gen, __ATOMIC_RELAXED, __HIP_MEMORY_SCOPE_AGENT);
    unsigned old = __hip_atomic_fetch_add(&b->cnt, 1u, __ATOMIC_ACQ_REL, __HIP_MEMORY_SCOPE_AGENT);
    if (old == NWG - 1) {
      __hip_atomic_store(&b->cnt, 0u, __ATOMIC_RELAXED, __HIP_MEMORY_SCOPE_AGENT);
      __hip_atomic_fetch_add(&b->gen, 1u, __ATOMIC_RELEASE, __HIP_MEMORY_SCOPE_AGENT);
    } else {
      while (__hip_atomic_load(&b->gen, __ATOMIC_ACQUIRE, __HIP_MEMORY_SCOPE_AGENT) == g)
        __builtin_amdgcn_s_sleep(2);
    }
  }
  __syncthreads();
}

__global__ void __launch_bounds__(TPB, 1)
lstm_persistent(const float* __restrict__ x,      // [S, I]
                const float* __restrict__ W_ih,   // [4H, I]
                const float* __restrict__ W_hh,   // [4H, H]
                const float* __restrict__ b_ih,   // [4H]
                const float* __restrict__ b_hh,   // [4H]
                const float* __restrict__ W_lin,  // [1, H]
                const float* __restrict__ b_lin,  // [1]
                float* __restrict__ out,          // [1]
                unsigned short* __restrict__ h_bf, // ws: 2*H bf16 (ping-pong)
                float* __restrict__ h_f32,         // ws: H floats (last step)
                Bar* __restrict__ bar) {
  __shared__ __align__(16) unsigned short h_lds[HDIM];    // 2 KB
  __shared__ __align__(16) unsigned short x_lds[IDIM];    // 128 B
  __shared__ __align__(16) unsigned short zero_blk[32];   // 64 B of zeros
  __shared__ float gates_lds[4 * JPW];
  __shared__ float c_lds[JPW];
  __shared__ float red[TPB];

  const int tid   = threadIdx.x;
  const int wv    = tid >> 5;
  const int lane  = tid & 31;
  const int n     = lane & 15;       // N within 16-wide tile
  const int half  = lane >> 4;       // K-half select
  const int wg    = blockIdx.x;
  const int gate  = wv >> 1;
  const int jhalf = wv & 1;
  const int row   = gate * HDIM + wg * JPW + jhalf * 16 + n;

  // ---- one-time: weight slices -> bf16 WMMA B-fragments, VGPR-resident ----
  v16bf bh[HDIM / 32];
#pragma unroll
  for (int kk = 0; kk < HDIM / 32; ++kk) {
    const float* src = W_hh + (size_t)row * HDIM + kk * 32 + half * 16;
    v16bf f;
#pragma unroll
    for (int e = 0; e < 16; ++e) f[e] = (__bf16)src[e];
    bh[kk] = f;
  }
  v16bf bx[IDIM / 32];
#pragma unroll
  for (int kk = 0; kk < IDIM / 32; ++kk) {
    const float* src = W_ih + (size_t)row * IDIM + kk * 32 + half * 16;
    v16bf f;
#pragma unroll
    for (int e = 0; e < 16; ++e) f[e] = (__bf16)src[e];
    bx[kk] = f;
  }

  // Bias pre-loaded into accumulator chain A, row M=0 (elem 0, lanes 0-15).
  v8f accA0, zacc;
#pragma unroll
  for (int e = 0; e < 8; ++e) { accA0[e] = 0.f; zacc[e] = 0.f; }
  if (half == 0) accA0[0] = b_ih[row] + b_hh[row];

  if (tid < JPW) c_lds[tid] = 0.f;
  if (tid < 32)  zero_blk[tid] = 0;   // zero page for inactive A lanes

  // Per-lane A-fragment source pointers: lanes 0/16 walk h/x data, all other
  // lanes sit on the zero block with stride 0 (reads stay in-bounds: +0..31B).
  const bool a_active = (n == 0);
  const unsigned short* hbase = a_active ? (h_lds + half * 16) : zero_blk;
  const unsigned short* xbase = a_active ? (x_lds + half * 16) : zero_blk;
  const int astep = a_active ? 32 : 0;   // ushorts per 32-wide K chunk

  // ------------------------------ time loop --------------------------------
  for (int t = 0; t < SEQ; ++t) {
    // Stage x_t (fp32 -> bf16) and h_{t-1} (ping-pong t&1) into LDS.
    if (tid < IDIM) x_lds[tid] = f2bf(x[(size_t)t * IDIM + tid]);
    {
      unsigned short* hsrc =
          const_cast<unsigned short*>(h_bf) + (t & 1) * HDIM + tid * 4;
#ifdef USE_ASYNC_LDS
      __builtin_amdgcn_global_load_async_to_lds_b64(
          (gv2i_p)hsrc, (lv2i_p)(h_lds + tid * 4), 0, 0);
      __builtin_amdgcn_s_wait_asynccnt(0);
#else
      *reinterpret_cast<uint2*>(h_lds + tid * 4) =
          *reinterpret_cast<const uint2*>(hsrc);
#endif
    }
    if (wv == 0 && t + 1 < SEQ) __builtin_prefetch(&x[(size_t)(t + 1) * IDIM], 0, 1);
    __syncthreads();

    // gates[row] = b + x_t @ W_ih[row,:] + h @ W_hh[row,:]
    // Two independent accumulator chains (even/odd chunks) + 2-deep A pipeline.
    v8f accA = accA0;
    v8f accB = zacc;
    {
      v16bf a0 = load_afrag(xbase);
      v16bf a1 = load_afrag(xbase + astep);
      accA = __builtin_amdgcn_wmma_f32_16x16x32_bf16(false, a0, false, bx[0],
                                                     (short)0, accA, false, false);
      accB = __builtin_amdgcn_wmma_f32_16x16x32_bf16(false, a1, false, bx[1],
                                                     (short)0, accB, false, false);
    }
    {
      const unsigned short* p = hbase;
      v16bf a0 = load_afrag(p); p += astep;
      v16bf a1 = load_afrag(p); p += astep;
#pragma unroll
      for (int kk = 0; kk < HDIM / 32; kk += 2) {
        v16bf a2 = (kk + 2 < HDIM / 32) ? load_afrag(p) : a0;  p += astep;
        accA = __builtin_amdgcn_wmma_f32_16x16x32_bf16(false, a0, false, bh[kk],
                                                       (short)0, accA, false, false);
        v16bf a3 = (kk + 3 < HDIM / 32) ? load_afrag(p) : a1;  p += astep;
        accB = __builtin_amdgcn_wmma_f32_16x16x32_bf16(false, a1, false, bh[kk + 1],
                                                       (short)0, accB, false, false);
        a0 = a2; a1 = a3;
      }
    }
    // D row M=0 lives in elem 0 on lanes 0-15; merge the two chains there.
    if (half == 0) gates_lds[gate * JPW + jhalf * 16 + n] = accA[0] + accB[0];
    __syncthreads();

    // Cell update for this WG's 32 hidden units.
    if (tid < JPW) {
      float gi = gates_lds[0 * JPW + tid];
      float gf = gates_lds[1 * JPW + tid];
      float gg = gates_lds[2 * JPW + tid];
      float go = gates_lds[3 * JPW + tid];
      float ii = 1.f / (1.f + __expf(-gi));
      float ff = 1.f / (1.f + __expf(-gf));
      float g2 = tanhf(gg);
      float oo = 1.f / (1.f + __expf(-go));
      float c  = ff * c_lds[tid] + ii * g2;
      c_lds[tid] = c;
      float h  = oo * tanhf(c);
      h_bf[((t + 1) & 1) * HDIM + wg * JPW + tid] = f2bf(h);
      if (t == SEQ - 1) h_f32[wg * JPW + tid] = h;
    }
    global_barrier(bar, tid);   // publish h_t to all 32 WGs
  }

  // ----------------------- final projection (WG 0) -------------------------
  if (wg == 0) {
    float p = 0.f;
#pragma unroll
    for (int u = 0; u < HDIM / TPB; ++u) {
      int j = u * TPB + tid;
      p += h_f32[j] * W_lin[j];
    }
    red[tid] = p;
    __syncthreads();
    for (int s2 = TPB / 2; s2 > 0; s2 >>= 1) {
      if (tid < s2) red[tid] += red[tid + s2];
      __syncthreads();
    }
    if (tid == 0) out[0] = red[0] + b_lin[0];
  }
}

// Re-initialize workspace every call (ws is poisoned once with 0xAA; stream
// ordering guarantees this runs before the main kernel -> deterministic).
__global__ void init_ws(unsigned short* h_bf, Bar* bar) {
  int tid = blockIdx.x * blockDim.x + threadIdx.x;
  if (tid == 0) { bar->cnt = 0u; bar->gen = 0u; }
  if (tid < 2 * HDIM) h_bf[tid] = 0;
}

extern "C" void kernel_launch(void* const* d_in, const int* in_sizes, int n_in,
                              void* d_out, int out_size, void* d_ws, size_t ws_size,
                              hipStream_t stream) {
  (void)in_sizes; (void)n_in; (void)out_size; (void)ws_size;
  const float* x     = (const float*)d_in[0];
  const float* W_ih  = (const float*)d_in[1];
  const float* W_hh  = (const float*)d_in[2];
  const float* b_ih  = (const float*)d_in[3];
  const float* b_hh  = (const float*)d_in[4];
  const float* W_lin = (const float*)d_in[5];
  const float* b_lin = (const float*)d_in[6];
  float* out = (float*)d_out;

  char* ws = (char*)d_ws;
  Bar* bar             = (Bar*)ws;                         // 8 B
  unsigned short* h_bf = (unsigned short*)(ws + 256);      // 2*H bf16 = 4 KB
  float* h_f32         = (float*)(ws + 256 + 4096);        // H floats

  hipLaunchKernelGGL(init_ws, dim3(8), dim3(256), 0, stream, h_bf, bar);
  hipLaunchKernelGGL(lstm_persistent, dim3(NWG), dim3(TPB), 0, stream,
                     x, W_ih, W_hh, b_ih, b_hh, W_lin, b_lin,
                     out, h_bf, h_f32, bar);
}
